// SparseNibblePPR_60224031424550
// MI455X (gfx1250) — matrix operand order repaired and checked by reading, not compiled, not run on hardware
//
#include <hip/hip_runtime.h>
#include <hip/hip_bf16.h>

typedef __bf16 bf16_t;
typedef __attribute__((ext_vector_type(16))) __bf16 v16bf;
typedef __attribute__((ext_vector_type(8)))  float  v8f;

#define F_DIM 128
#define D_DIM 64
#define K_NBR 128

// ---------------------------------------------------------------------------
// Phase 1: enc[N, 64] = X[N,128] @ W[128,64] + b   via v_wmma_f32_16x16x32_bf16
//
// One wave computes a 16-row x 64-col tile of enc (4 col tiles x 4 K-chunks
// = 16 WMMAs). W (32 KB fp32) is staged ONCE per block into LDS, converted to
// bf16 and pre-swizzled into this kernel's B-fragment VGPR layout (contiguous
// k = 16*half + j, same as A), so the inner loop reads each fragment with a single
// 32-byte DS load instead of 16 scalar global loads + converts.
// ---------------------------------------------------------------------------
__global__ void __launch_bounds__(128)
ppr_encode_wmma(const float* __restrict__ X, const float* __restrict__ W,
                const float* __restrict__ bias, float* __restrict__ enc, int N)
{
    const int lane = threadIdx.x & 31;
    const int waveInBlk = threadIdx.x >> 5;     // 0..3
    const int half = lane >> 4;                 // 0: lanes 0-15, 1: lanes 16-31
    const int mn   = lane & 15;                 // A row / B-C-D column (in tile)

    // ---- Stage W into LDS as bf16 B-fragments.
    // Fragment (kc, t, lane) = 16 bf16 (32 B): elem j -> W[kc*32+half*16+j][t*16+mn]
    __shared__ alignas(32) bf16_t wfrag[4 * 4 * 32 * 16];   // 16 KB
    {
        const int kc = waveInBlk;               // each wave stages one K-chunk
        #pragma unroll
        for (int t = 0; t < 4; ++t) {
            const int fragBase = ((kc * 4 + t) * 32 + lane) * 16;
            const float* __restrict__ wcol =
                W + (long)(kc * 32 + half * 16) * D_DIM + (t * 16 + mn);
            #pragma unroll
            for (int j = 0; j < 16; ++j)
                wfrag[fragBase + j] = (bf16_t)wcol[j * D_DIM];
        }
    }
    __syncthreads();

    const int wavesPerBlk = blockDim.x >> 5;
    const int tile = blockIdx.x * wavesPerBlk + waveInBlk;   // 16-row tile id
    const int row0 = tile * 16;
    if (row0 >= N) return;

    const long arow = (long)(row0 + mn);
    const float* __restrict__ xrow = X + arow * F_DIM;

    v8f acc[4];
    #pragma unroll
    for (int t = 0; t < 4; ++t)
        #pragma unroll
        for (int r = 0; r < 8; ++r) acc[t][r] = 0.0f;

    #pragma unroll
    for (int kc = 0; kc < 4; ++kc) {
        const int kbase = kc * 32;

        // ---- A fragment (16x32 bf16): elem j -> K=kbase+half*16+j.
        //      One contiguous 16-float run.
        v16bf afrag;
        const float4 xa0 = *(const float4*)(xrow + kbase + half * 16);
        const float4 xa1 = *(const float4*)(xrow + kbase + half * 16 + 4);
        const float4 xb0 = *(const float4*)(xrow + kbase + half * 16 + 8);
        const float4 xb1 = *(const float4*)(xrow + kbase + half * 16 + 12);
        afrag[0]  = (bf16_t)xa0.x; afrag[1]  = (bf16_t)xa0.y;
        afrag[2]  = (bf16_t)xa0.z; afrag[3]  = (bf16_t)xa0.w;
        afrag[4]  = (bf16_t)xa1.x; afrag[5]  = (bf16_t)xa1.y;
        afrag[6]  = (bf16_t)xa1.z; afrag[7]  = (bf16_t)xa1.w;
        afrag[8]  = (bf16_t)xb0.x; afrag[9]  = (bf16_t)xb0.y;
        afrag[10] = (bf16_t)xb0.z; afrag[11] = (bf16_t)xb0.w;
        afrag[12] = (bf16_t)xb1.x; afrag[13] = (bf16_t)xb1.y;
        afrag[14] = (bf16_t)xb1.z; afrag[15] = (bf16_t)xb1.w;

        #pragma unroll
        for (int t = 0; t < 4; ++t) {
            const v16bf bfrag =
                *(const v16bf*)&wfrag[((kc * 4 + t) * 32 + lane) * 16];
            acc[t] = __builtin_amdgcn_wmma_f32_16x16x32_bf16(
                /*neg_a=*/false, afrag, /*neg_b=*/false, bfrag,
                /*c_mod=*/(short)0, acc[t], /*reuse_a=*/false, /*reuse_b=*/false);
        }
    }

    // ---- Store: VGPR r -> row row0 + r + 8*half, column t*16 + mn
    #pragma unroll
    for (int t = 0; t < 4; ++t) {
        const int col = t * 16 + mn;
        const float bv = bias[col];
        #pragma unroll
        for (int r = 0; r < 8; ++r)
            enc[(long)(row0 + r + 8 * half) * D_DIM + col] = acc[t][r] + bv;
    }
}

// ---------------------------------------------------------------------------
// Phase 2: out[b, :] = sum_k values[idx[b],k] * enc[indices[idx[b],k], :]
// One wave32 per seed; each lane owns 2 of the 64 output columns (float2),
// so every k-step gathers one full 256 B enc row across the wave. enc
// (25.6 MB) is L2-resident (192 MB L2), so the 512 MB of gathers never touch
// HBM. Wave-uniform neighbor ids / weights are fetched 4-at-a-time (16 B).
// ---------------------------------------------------------------------------
__global__ void __launch_bounds__(256)
ppr_gather_wsum(const int* __restrict__ idx, const int* __restrict__ indices,
                const float* __restrict__ values, const float* __restrict__ enc,
                float* __restrict__ out, int B)
{
    const int lane = threadIdx.x & 31;
    const int wave = (int)((blockIdx.x * blockDim.x + threadIdx.x) >> 5);
    if (wave >= B) return;

    const long seed = (long)idx[wave];
    const int*   __restrict__ ind = indices + seed * K_NBR;   // 512 B aligned
    const float* __restrict__ val = values  + seed * K_NBR;   // 512 B aligned

    float2 acc; acc.x = 0.0f; acc.y = 0.0f;
    const int coff = lane * 2;

    #pragma unroll 2
    for (int k = 0; k < K_NBR; k += 4) {
        const int4   nb = *(const int4*)  (ind + k);   // wave-uniform, 16 B
        const float4 vv = *(const float4*)(val + k);   // wave-uniform, 16 B

        float2 e0 = *(const float2*)(enc + (long)nb.x * D_DIM + coff);
        float2 e1 = *(const float2*)(enc + (long)nb.y * D_DIM + coff);
        float2 e2 = *(const float2*)(enc + (long)nb.z * D_DIM + coff);
        float2 e3 = *(const float2*)(enc + (long)nb.w * D_DIM + coff);

        acc.x = fmaf(vv.x, e0.x, acc.x);  acc.y = fmaf(vv.x, e0.y, acc.y);
        acc.x = fmaf(vv.y, e1.x, acc.x);  acc.y = fmaf(vv.y, e1.y, acc.y);
        acc.x = fmaf(vv.z, e2.x, acc.x);  acc.y = fmaf(vv.z, e2.y, acc.y);
        acc.x = fmaf(vv.w, e3.x, acc.x);  acc.y = fmaf(vv.w, e3.y, acc.y);
    }

    *(float2*)(out + (long)wave * D_DIM + coff) = acc;
}

extern "C" void kernel_launch(void* const* d_in, const int* in_sizes, int n_in,
                              void* d_out, int out_size, void* d_ws, size_t ws_size,
                              hipStream_t stream)
{
    const float* X       = (const float*)d_in[0];   // [N, 128]
    const int*   idx     = (const int*)  d_in[1];   // [B]
    const int*   indices = (const int*)  d_in[2];   // [N, 128]
    const float* values  = (const float*)d_in[3];   // [N, 128]
    const float* W       = (const float*)d_in[4];   // [128, 64]
    const float* bias    = (const float*)d_in[5];   // [64]

    const int N = in_sizes[0] / F_DIM;              // 100000
    const int B = in_sizes[1];                      // 16384

    float* enc = (float*)d_ws;                      // [N, 64] scratch (25.6 MB)
    float* out = (float*)d_out;                     // [B, 64]

    // Phase 1: WMMA encode. 16-row tiles, 4 waves (128 thr) per block.
    const int tiles   = (N + 15) / 16;
    const int blocks1 = (tiles + 3) / 4;
    ppr_encode_wmma<<<blocks1, 128, 0, stream>>>(X, W, bias, enc, N);

    // Phase 2: gather + weighted sum. 1 wave per seed, 8 waves per block.
    const int blocks2 = (B + 7) / 8;
    ppr_gather_wsum<<<blocks2, 256, 0, stream>>>(idx, indices, values, enc, out, B);
}
